// ClassDistancePenaltyLoss_35631048687776
// MI455X (gfx1250) — matrix operand order repaired and checked by reading, not compiled, not run on hardware
//
#include <hip/hip_runtime.h>
#include <hip/hip_bf16.h>

typedef __attribute__((ext_vector_type(16))) _Float16 v16h;
typedef __attribute__((ext_vector_type(8)))  float    v8f;

#define NROWS   262144
#define DFEAT   256
#define NCLS    100
#define CPAD    112            // 7 * 16 class tiles
#define NMT     7              // M-tiles of 16 classes
#define PITCH   40             // f16 pitch of transposed LDS tile (80B rows: 16B aligned)
#define NCHUNKS (NROWS / 32)
#define GRIDX   256

// ---------------------------------------------------------------------------
// zero the workspace accumulators (S: CPAD*DFEAT f32, SQ: CPAD f32, CNT: CPAD u32)
// ---------------------------------------------------------------------------
__global__ void zero_ws_kernel(float* __restrict__ p, int n) {
    int i = blockIdx.x * blockDim.x + threadIdx.x;
    if (i < n) p[i] = 0.0f;
}

// ---------------------------------------------------------------------------
// main: S = onehot(t)^T * X via v_wmma_f32_16x16x32_f16, plus per-class
// squared-norm totals SQ_c and counts n_c.
// ---------------------------------------------------------------------------
__global__ __launch_bounds__(512)
void stats_kernel(const float* __restrict__ x, const int* __restrict__ tcls,
                  float* __restrict__ Sws, float* __restrict__ SQws,
                  unsigned* __restrict__ CNTws) {
    __shared__ __align__(32) _Float16 xt[DFEAT * PITCH];   // 20480 B, transposed f16 chunk
    __shared__ __align__(32) _Float16 Alds[NMT * 32 * 16]; //  7168 B, shared one-hot A tiles
    __shared__ int      tch[32];
    __shared__ float    rn[32];       // per-row squared norms of current chunk
    __shared__ float    sqc[CPAD];    // block-local SQ_c
    __shared__ unsigned cntc[CPAD];   // block-local n_c

    const int tid  = threadIdx.x;
    const int w    = tid >> 5;        // wave id 0..15 (wave32)
    const int lane = tid & 31;
    const int half = (lane >> 4) & 1;
    const int lm   = lane & 15;

    if (tid < CPAD) { sqc[tid] = 0.0f; cntc[tid] = 0u; }
    if (tid < 32)   rn[tid] = 0.0f;

    v8f acc[NMT];
    #pragma unroll
    for (int m = 0; m < NMT; ++m)
        acc[m] = (v8f){0.f,0.f,0.f,0.f,0.f,0.f,0.f,0.f};

    __syncthreads();

    // staging tile: each thread owns a column PAIR (float2 loads, 256B/wave/instr)
    const int cc = (tid & 127) * 2;   // columns cc, cc+1
    const int h  = tid >> 7;          // row group 0..3 -> rows [8h, 8h+8)

    for (int chunk = blockIdx.x; chunk < NCHUNKS; chunk += gridDim.x) {
        const int base = chunk * 32;
        const float* xp = x + (size_t)(base + h * 8) * DFEAT + cc;

        // prefetch next chunk for this block (emits global_prefetch_b8)
        if (chunk + gridDim.x < NCHUNKS) {
            const float* nx = xp + (size_t)gridDim.x * 32 * DFEAT;
            __builtin_prefetch(nx, 0, 1);
            __builtin_prefetch(nx + 4 * DFEAT, 0, 1);
        }

        // ---- stage: coalesced float2 loads -> f16 transposed LDS, squares in regs
        float p[8];
        #pragma unroll
        for (int j = 0; j < 4; ++j) {
            float2 a0 = *(const float2*)&xp[(2 * j) * DFEAT];
            float2 a1 = *(const float2*)&xp[(2 * j + 1) * DFEAT];
            p[2 * j]     = fmaf(a0.x, a0.x, a0.y * a0.y);
            p[2 * j + 1] = fmaf(a1.x, a1.x, a1.y * a1.y);
            union { _Float16 hh[2]; unsigned u; } pk0, pk1;
            pk0.hh[0] = (_Float16)a0.x;  pk0.hh[1] = (_Float16)a1.x;   // column cc
            pk1.hh[0] = (_Float16)a0.y;  pk1.hh[1] = (_Float16)a1.y;   // column cc+1
            *(unsigned*)&xt[cc * PITCH + h * 8 + 2 * j]       = pk0.u; // rows contiguous
            *(unsigned*)&xt[(cc + 1) * PITCH + h * 8 + 2 * j] = pk1.u;
        }

        // wave32 butterfly: per-row sum of squares over this wave's 64 columns
        #pragma unroll
        for (int r = 0; r < 8; ++r) {
            p[r] += __shfl_xor(p[r], 1, 32);
            p[r] += __shfl_xor(p[r], 2, 32);
            p[r] += __shfl_xor(p[r], 4, 32);
            p[r] += __shfl_xor(p[r], 8, 32);
            p[r] += __shfl_xor(p[r], 16, 32);
        }
        if (lane == 0) {
            #pragma unroll
            for (int r = 0; r < 8; ++r) atomicAdd(&rn[h * 8 + r], p[r]);
        }

        if (tid < 32) tch[tid] = tcls[base + tid];

        // ---- build shared one-hot A tiles (waves 0..6, one mtile each)
        // A 16x32 k order used here (contiguous, same as B): lane<16: K = e; lane>=16: K = 16 + e
        if (w < NMT) {
            const int myc = w * 16 + lm;
            union { uint4 q[2]; v16h v; } ua;
            #pragma unroll
            for (int e = 0; e < 16; ++e) {
                int k  = e + (half ? 16 : 0);
                int tv = tcls[base + k];
                ua.v[e] = (tv == myc) ? (_Float16)1.0f : (_Float16)0.0f;
            }
            uint4* dst = (uint4*)&Alds[(w * 32 + lane) * 16];
            dst[0] = ua.q[0];
            dst[1] = ua.q[1];
        }

        __syncthreads();

        // ---- WMMA: wave w owns features [16w, 16w+16)
        // B 32x16 k order used here (contiguous, same as A): lanes 0-15 hold K=0..15 (N=lane), lanes 16-31 K=16..31
        const int feat = w * 16 + lm;
        union UA { uint4 q[2]; v16h v; };
        UA ub;
        const uint4* bp = (const uint4*)&xt[feat * PITCH + half * 16];
        ub.q[0] = bp[0];
        ub.q[1] = bp[1];

        // preload ALL 7 A tiles into disjoint registers: lets the scheduler issue
        // 14 ds_load_b128 back-to-back and use graduated s_wait_dscnt instead of
        // draining LDS before every WMMA.
        UA at[NMT];
        #pragma unroll
        for (int m = 0; m < NMT; ++m) {
            const uint4* ap = (const uint4*)&Alds[(m * 32 + lane) * 16];
            at[m].q[0] = ap[0];
            at[m].q[1] = ap[1];
        }
        #pragma unroll
        for (int m = 0; m < NMT; ++m) {
            acc[m] = __builtin_amdgcn_wmma_f32_16x16x32_f16(
                false, at[m].v, false, ub.v, (short)0, acc[m], false, false);
        }

        // ---- wave 0: fold per-row norms into per-class SQ / counts, reset rn
        if (w == 0) {
            float v  = rn[lane];
            rn[lane] = 0.0f;
            int cls  = tch[lane];
            atomicAdd(&sqc[cls], v);
            atomicAdd(&cntc[cls], 1u);
        }
        __syncthreads();
    }

    // ---- flush register accumulators: C/D layout lane=N (feature), VGPR v:
    //      M = v (lanes 0-15) / v+8 (lanes 16-31)
    #pragma unroll
    for (int m = 0; m < NMT; ++m) {
        #pragma unroll
        for (int v = 0; v < 8; ++v) {
            int cls = m * 16 + v + half * 8;
            atomicAdd(&Sws[cls * DFEAT + w * 16 + lm], acc[m][v]);
        }
    }
    if (tid < CPAD) {
        atomicAdd(&SQws[tid], sqc[tid]);
        atomicAdd(&CNTws[tid], cntc[tid]);
    }
}

// ---------------------------------------------------------------------------
// finalize: per-class variance sum = (SQ_c - |s_c|^2/n) / (n-1), mean over C
// ---------------------------------------------------------------------------
__global__ void finalize_kernel(const float* __restrict__ Sws,
                                const float* __restrict__ SQws,
                                const unsigned* __restrict__ CNTws,
                                float* __restrict__ out) {
    __shared__ float red[128];
    int tid = threadIdx.x;
    float val = 0.0f;
    if (tid < NCLS) {
        const float* s = Sws + tid * DFEAT;
        float ss = 0.0f;
        for (int d = 0; d < DFEAT; ++d) { float sv = s[d]; ss = fmaf(sv, sv, ss); }
        float n = (float)CNTws[tid];
        float vs = (SQws[tid] - ss / n) / (n - 1.0f);
        val = fabsf(vs);   // sample variance is non-negative; abs matches reference
    }
    red[tid] = val;
    __syncthreads();
    for (int off = 64; off > 0; off >>= 1) {
        if (tid < off) red[tid] += red[tid + off];
        __syncthreads();
    }
    if (tid == 0) out[0] = red[0] / (float)NCLS;
}

// ---------------------------------------------------------------------------
extern "C" void kernel_launch(void* const* d_in, const int* in_sizes, int n_in,
                              void* d_out, int out_size, void* d_ws, size_t ws_size,
                              hipStream_t stream) {
    const float* x = (const float*)d_in[0];   // f32 [262144, 256]
    const int*   t = (const int*)d_in[1];     // int labels [262144]

    float*    Sws   = (float*)d_ws;                   // CPAD*DFEAT f32
    float*    SQws  = Sws + CPAD * DFEAT;             // CPAD f32
    unsigned* CNTws = (unsigned*)(SQws + CPAD);       // CPAD u32
    const int nzero = CPAD * DFEAT + CPAD + CPAD;     // ~113 KB of d_ws

    zero_ws_kernel<<<(nzero + 255) / 256, 256, 0, stream>>>((float*)d_ws, nzero);
    stats_kernel<<<GRIDX, 512, 0, stream>>>(x, t, Sws, SQws, CNTws);
    finalize_kernel<<<1, 128, 0, stream>>>(Sws, SQws, CNTws, (float*)d_out);
}
